// PyTorchModel_18305150615594
// MI455X (gfx1250) — matrix operand order repaired and checked by reading, not compiled, not run on hardware
//
#include <hip/hip_runtime.h>

typedef __attribute__((ext_vector_type(16))) _Float16 v16h;
typedef __attribute__((ext_vector_type(8)))  float    v8f;

union frag16 { unsigned u[8]; v16h h; };

__device__ __forceinline__ unsigned pack2h(float a, float b) {
    union { _Float16 h[2]; unsigned u; } x;
    x.h[0] = (_Float16)a; x.h[1] = (_Float16)b;
    return x.u;
}

// Branch-free tanh: tanh(x) = 1 - 2/(exp2(2*log2(e)*x) + 1).
// Lowers to v_exp_f32 + v_rcp_f32 + a few VALU; saturates correctly at +-1.
__device__ __forceinline__ float fast_tanh(float x) {
    const float e = __builtin_amdgcn_exp2f(x * 2.8853900817779268f);
    return 1.0f - 2.0f * __builtin_amdgcn_rcpf(e + 1.0f);
}

// Sizes (from reference): B=32768, S=16, E=16, F=16, P=32, HID=128, X=P+F=48 (pad K->64)
#define NB     32768
#define NS     16
#define NE     16
#define NF     16
#define NP     32
#define NH     128
#define ROWS   256
#define NTHR   256

// ---------------------------------------------------------------------------
// Prep: repack W0 (E,48,128) and W1 (E,128,32) fp32 -> f16 WMMA B-fragment
// layout in d_ws, so the main kernel's per-lane loads are contiguous b128s.
// B packing used here (16x16x32 f16; A uses the same k order): lane L -> column n = L%16, g = L>>4,
// dword j holds halves K = chunkBase + 16*g + 2*j, +1.
// w0f: [e][t8(8 n-tiles)][c(2 k-chunks)][lane][j]  = 65536 dwords (K>=48 zeroed)
// w1f: [e][t2(2 n-tiles)][c(4 k-chunks)][lane][j]  = 32768 dwords
// ---------------------------------------------------------------------------
__global__ void prep_weights(const float* __restrict__ W0,
                             const float* __restrict__ W1,
                             unsigned* __restrict__ w0f,
                             unsigned* __restrict__ w1f) {
    const int idx = blockIdx.x * blockDim.x + threadIdx.x;
    if (idx < 65536) {
        const int j  = idx & 7;
        const int L  = (idx >> 3) & 31;
        const int c  = (idx >> 8) & 1;
        const int t8 = (idx >> 9) & 7;
        const int e  = (idx >> 12) & 15;
        const int g  = L >> 4;
        const int n  = t8 * 16 + (L & 15);
        const int k  = c * 32 + 16 * g + 2 * j;
        unsigned v = 0u;
        if (k + 1 < 48) {  // real K range is [0,48); rest is zero padding
            v = pack2h(W0[(e * 48 + k) * NH + n], W0[(e * 48 + k + 1) * NH + n]);
        }
        w0f[idx] = v;
    }
    if (idx < 32768) {
        const int j  = idx & 7;
        const int L  = (idx >> 3) & 31;
        const int c  = (idx >> 8) & 3;
        const int t2 = (idx >> 10) & 1;
        const int e  = (idx >> 11) & 15;
        const int g  = L >> 4;
        const int n  = t2 * 16 + (L & 15);
        const int k  = c * 32 + 16 * g + 2 * j;  // < 128 always
        w1f[idx] = pack2h(W1[(e * NH + k) * NP + n], W1[(e * NH + k + 1) * NP + n]);
    }
}

// ---------------------------------------------------------------------------
// Main recurrent MoE kernel. One block = 256 rows, 8 waves.
// Per step: build x=[p|feat] (f16, LDS), counting-sort active rows by expert
// into padded 16-row tiles, each tile -> WMMA MLP for its expert only.
// ---------------------------------------------------------------------------
__launch_bounds__(NTHR)
__global__ void moe_rnn_kernel(const int*   __restrict__ ids,
                               const float* __restrict__ feats,
                               const float* __restrict__ p_in,
                               const int*   __restrict__ seq_len,
                               const float* __restrict__ b0,
                               const float* __restrict__ b1,
                               const unsigned* __restrict__ w0f,
                               const unsigned* __restrict__ w1f,
                               float* __restrict__ out) {
    __shared__ float    pbuf[ROWS * NP];        // 32 KB  persistent p
    __shared__ _Float16 xh[ROWS * 64];          // 32 KB  [p|feat|0] rows, K padded to 64
    __shared__ _Float16 hsc[8 * 16 * NH];       // 32 KB  per-wave h scratch (16x128 f16)
    __shared__ float    b0s[NE * NH];           // 8 KB   biases staged in LDS
    __shared__ float    b1s[NE * NP];           // 2 KB
    __shared__ int      cnt[NE];
    __shared__ int      startE[NE];
    __shared__ short    tileExpert[32];
    __shared__ short    orderBuf[512];          // padded slot -> row (-1 = pad)
    __shared__ int      Tsh;

    const int t    = threadIdx.x;
    const int r0   = blockIdx.x * ROWS;
    const int lane = t & 31;
    const int wave = t >> 5;
    const int m16  = lane & 15;
    const int g    = lane >> 4;

    for (int i = t; i < ROWS * NP; i += NTHR) pbuf[i] = p_in[r0 * NP + i];
    for (int i = t; i < NE * NH; i += NTHR) b0s[i] = b0[i];
    for (int i = t; i < NE * NP; i += NTHR) b1s[i] = b1[i];
    const int slen = seq_len[r0 + t];
    __syncthreads();

    #pragma unroll 1
    for (int n = 0; n < NS; ++n) {
        // ---- build x row for my row: [p(32) | feat(16) | zeros(16)] in f16
        {
            _Float16* xr = xh + t * 64;
            #pragma unroll
            for (int k = 0; k < NP; ++k) xr[k] = (_Float16)pbuf[t * NP + k];
            const float* f = feats + ((size_t)(r0 + t) * NS + n) * NF;
            #pragma unroll
            for (int k = 0; k < NF; ++k) xr[NP + k] = (_Float16)f[k];
            #pragma unroll
            for (int k = 48; k < 64; ++k) xr[k] = (_Float16)0.f;
        }
        if (t < NE) cnt[t] = 0;
        for (int i = t; i < 512; i += NTHR) orderBuf[i] = (short)-1;
        __syncthreads();

        const int  myid   = ids[(r0 + t) * NS + n];
        const bool active = (n < slen);
        int rank = 0;
        if (active) rank = atomicAdd(&cnt[myid], 1);
        __syncthreads();

        if (t == 0) {  // serial 16-element prefix over padded bucket sizes
            int run = 0;
            for (int e = 0; e < NE; ++e) {
                startE[e] = run;
                const int pc = (cnt[e] + 15) & ~15;
                for (int q = 0; q < (pc >> 4); ++q) tileExpert[(run >> 4) + q] = (short)e;
                run += pc;
            }
            Tsh = run >> 4;
        }
        __syncthreads();
        if (active) orderBuf[startE[myid] + rank] = (short)t;
        __syncthreads();

        if (n + 1 < NS)  // gfx1250 global_prefetch_b8 for next step's features
            __builtin_prefetch(feats + ((size_t)(r0 + t) * NS + n + 1) * NF, 0, 0);

        const int T = Tsh;
        #pragma unroll 1
        for (int tile = wave; tile < T; tile += 8) {   // wave-uniform loop
            const int e = tileExpert[tile];

            // A fragments (layer 0), 2 K-chunks; A uses the same contiguous k order as the B packing: lane=M, pairs at
            // K = 16g+2j. Chunk1 K>=48 -> zero.
            frag16 a0, a1;
            {
                const int row = orderBuf[tile * 16 + m16];
                const unsigned* xr = (row >= 0) ? (const unsigned*)(xh + row * 64) : (const unsigned*)xh;
                const bool live = (row >= 0);
                #pragma unroll
                for (int j = 0; j < 8; ++j) {
                    const int k0 = 16 * g + 2 * j;
                    a0.u[j] = live ? xr[k0 >> 1] : 0u;
                    const int k1 = 32 + k0;
                    a1.u[j] = (live && k1 < 48) ? xr[k1 >> 1] : 0u;
                }
            }

            _Float16* hs = hsc + wave * (16 * NH);
            const unsigned* w0e = w0f + e * 4096;

            // ---- layer 0: h = tanh(x @ W0[e] + b0[e]), 8 N-tiles of 16
            #pragma unroll
            for (int t8 = 0; t8 < 8; ++t8) {
                const float bias = b0s[e * NH + t8 * 16 + m16];  // C/D: lane=N
                v8f acc;
                #pragma unroll
                for (int j = 0; j < 8; ++j) acc[j] = bias;
                frag16 bfr;
                {
                    const uint4* bp = (const uint4*)(w0e + t8 * 512 + lane * 8);
                    uint4 x0 = bp[0], x1 = bp[1];
                    bfr.u[0] = x0.x; bfr.u[1] = x0.y; bfr.u[2] = x0.z; bfr.u[3] = x0.w;
                    bfr.u[4] = x1.x; bfr.u[5] = x1.y; bfr.u[6] = x1.z; bfr.u[7] = x1.w;
                }
                acc = __builtin_amdgcn_wmma_f32_16x16x32_f16(false, a0.h, false, bfr.h,
                                                             (short)0, acc, false, false);
                {
                    const uint4* bp = (const uint4*)(w0e + t8 * 512 + 256 + lane * 8);
                    uint4 x0 = bp[0], x1 = bp[1];
                    bfr.u[0] = x0.x; bfr.u[1] = x0.y; bfr.u[2] = x0.z; bfr.u[3] = x0.w;
                    bfr.u[4] = x1.x; bfr.u[5] = x1.y; bfr.u[6] = x1.z; bfr.u[7] = x1.w;
                }
                acc = __builtin_amdgcn_wmma_f32_16x16x32_f16(false, a1.h, false, bfr.h,
                                                             (short)0, acc, false, false);
                #pragma unroll
                for (int j = 0; j < 8; ++j) {   // D: lane=N, VGPR j -> M=8g+j
                    const int m = 8 * g + j;
                    hs[m * NH + t8 * 16 + m16] = (_Float16)fast_tanh(acc[j]);
                }
            }

            // A fragments (layer 1) from per-wave h scratch (wave-internal dep)
            frag16 ha[4];
            {
                const unsigned* hr = (const unsigned*)(hs + m16 * NH);
                #pragma unroll
                for (int c = 0; c < 4; ++c) {
                    #pragma unroll
                    for (int j = 0; j < 8; ++j) {
                        const int k0 = c * 32 + 16 * g + 2 * j;
                        ha[c].u[j] = hr[k0 >> 1];
                    }
                }
            }

            // ---- layer 1: o = h @ W1[e] + b1[e], 2 N-tiles x 4 K-chunks
            const unsigned* w1e = w1f + e * 2048;
            #pragma unroll
            for (int t2 = 0; t2 < 2; ++t2) {
                const float bias = b1s[e * NP + t2 * 16 + m16];
                v8f acc;
                #pragma unroll
                for (int j = 0; j < 8; ++j) acc[j] = bias;
                #pragma unroll
                for (int c = 0; c < 4; ++c) {
                    frag16 bfr;
                    const uint4* bp = (const uint4*)(w1e + t2 * 1024 + c * 256 + lane * 8);
                    uint4 x0 = bp[0], x1 = bp[1];
                    bfr.u[0] = x0.x; bfr.u[1] = x0.y; bfr.u[2] = x0.z; bfr.u[3] = x0.w;
                    bfr.u[4] = x1.x; bfr.u[5] = x1.y; bfr.u[6] = x1.z; bfr.u[7] = x1.w;
                    acc = __builtin_amdgcn_wmma_f32_16x16x32_f16(false, ha[c].h, false, bfr.h,
                                                                 (short)0, acc, false, false);
                }
                #pragma unroll
                for (int j = 0; j < 8; ++j) {   // scatter new p rows
                    const int m   = 8 * g + j;
                    const int row = orderBuf[tile * 16 + m];
                    if (row >= 0) pbuf[row * NP + t2 * 16 + m16] = acc[j];
                }
            }
        }
        __syncthreads();  // p updates visible before next step's x build
    }

    for (int i = t; i < ROWS * NP; i += NTHR) {
        const float v = pbuf[i];
        out[r0 * NP + i] = v > 0.f ? v : 0.f;
    }
}

extern "C" void kernel_launch(void* const* d_in, const int* in_sizes, int n_in,
                              void* d_out, int out_size, void* d_ws, size_t ws_size,
                              hipStream_t stream) {
    const int*   mod_id_seq   = (const int*)d_in[0];
    const float* mod_feat_seq = (const float*)d_in[1];
    const float* p_in         = (const float*)d_in[2];
    const int*   seq_len      = (const int*)d_in[3];
    const float* W0           = (const float*)d_in[4];
    const float* b0           = (const float*)d_in[5];
    const float* W1           = (const float*)d_in[6];
    const float* b1           = (const float*)d_in[7];
    float*       out          = (float*)d_out;

    unsigned* w0f = (unsigned*)d_ws;          // 65536 dwords = 256 KB
    unsigned* w1f = w0f + 65536;              // 32768 dwords = 128 KB

    prep_weights<<<256, 256, 0, stream>>>(W0, W1, w0f, w1f);
    moe_rnn_kernel<<<NB / ROWS, NTHR, 0, stream>>>(mod_id_seq, mod_feat_seq, p_in,
                                                   seq_len, b0, b1, w0f, w1f, out);
}
